// FasterRCNN_36567351558371
// MI455X (gfx1250) — matrix operand order repaired and checked by reading, not compiled, not run on hardware
//
#include <hip/hip_runtime.h>
#include <hip/hip_bf16.h>

// ---------- types ----------
typedef __attribute__((ext_vector_type(16))) __bf16 bf16x16;
typedef __attribute__((ext_vector_type(8)))  float  f32x8;

union Frag { uint4 q[2]; bf16x16 v; };

// ---------- problem constants ----------
#define Hh 40
#define Ww 40
#define CIN 2048
#define CMID 256
#define KDIM (9 * CIN)            // 18432
#define NPIX (Hh * Ww)            // 1600
#define NANCH (NPIX * 5)          // 8000
#define SORTN 8192
#define MAXDET 300
#define KSLICES 4
#define STAGES_TOTAL (KDIM / 64)  // 288 stages of 64 K-values
#define STAGES_PER_SLICE (STAGES_TOTAL / KSLICES)   // 72

// workspace byte offsets (all 16B aligned)
#define OFS_FEATPAD 0u                       // [42*42][2048] bf16 = 7,225,344
#define OFS_WPACK   7225344u                 // [576][256][32] bf16 = 9,437,184
#define OFS_X       16662528u                // [1600][256] f32 = 1,638,400
#define OFS_LOGITS  18300928u                // [1600][10] f32
#define OFS_DELTAS  18364928u                // [1600][20] f32
#define OFS_SCORES  18492928u                // [8000] f32
#define OFS_BOXES   18524928u                // [8000][4] f32
#define OFS_SBOXES  18652928u                // [8000][4] f32 (sorted)
#define OFS_KEEP    18780928u                // [8000] i32
#define OFS_FC      18812928u                // [300][1024] f32
#define OFS_XPART   20041728u                // 4 x [1600][256] f32 = 6,553,600

__device__ __forceinline__ unsigned short f2bf(float f) {
  unsigned int u = __float_as_uint(f);
  unsigned int r = u + 0x7FFFu + ((u >> 16) & 1u);
  return (unsigned short)(r >> 16);
}

// ---------- 1) pad + convert features to bf16 [42][42][2048] ----------
__global__ void k_pad_features(const float* __restrict__ feat,
                               unsigned short* __restrict__ fpad) {
  int idx = blockIdx.x * blockDim.x + threadIdx.x;
  const int total = 42 * 42 * CIN;
  if (idx >= total) return;
  int c = idx & (CIN - 1);
  int p = idx >> 11;
  int px = p % 42, py = p / 42;
  float v = 0.f;
  if (py >= 1 && py <= Hh && px >= 1 && px <= Ww)
    v = feat[(((py - 1) * Ww) + (px - 1)) * CIN + c];
  fpad[idx] = f2bf(v);
}

// ---------- 2) repack conv_w (3,3,2048,256) f32 -> bf16 [K/32][256][32] ----------
__global__ void k_pack_weights(const float* __restrict__ w,
                               unsigned short* __restrict__ wp) {
  int idx = blockIdx.x * blockDim.x + threadIdx.x;
  const int total = KDIM * CMID;
  if (idx >= total) return;
  int k = idx / CMID, n = idx % CMID;
  wp[(k >> 5) * (CMID * 32) + n * 32 + (k & 31)] = f2bf(w[idx]);
}

// ---------- 3) implicit-GEMM 3x3 conv: async-LDS B panel + WMMA bf16 ----------
// Grid: 200 blocks x 256 threads. Block b: K-slice = b/50, M-pair = b%50.
// Wave (8/block): local M-tile = wv>>2 (2 tiles of 16 rows), N-quad = wv&3 (64 cols).
// Each wave: 16x64 output, 4 f32x8 accumulators, K loop over its slice.
__global__ __launch_bounds__(256) void k_conv3x3_wmma(
    const unsigned short* __restrict__ fpad,
    const unsigned short* __restrict__ wp,
    float* __restrict__ xpart) {
  __shared__ __align__(16) unsigned short Bsm[2][16384];   // 2 x 32KB double buffer

  int t    = threadIdx.x;
  int lane = t & 31;
  int wv   = t >> 5;
  int kslice = blockIdx.x / 50;
  int mb     = blockIdx.x % 50;
  int mt = mb * 2 + (wv >> 2);
  int nq = wv & 3;
  int lm   = lane & 15;
  int half = lane >> 4;                 // A: 0 -> K 0..7/16..23, 1 -> K 8..15/24..31
  int row  = mt * 16 + lm;
  int y = row / Ww, x = row % Ww;

  unsigned ldsb[2];
  ldsb[0] = (unsigned)(unsigned long long)(&Bsm[0][0]);
  ldsb[1] = (unsigned)(unsigned long long)(&Bsm[1][0]);
  unsigned long long wbase = (unsigned long long)wp;

  const int s0 = kslice * STAGES_PER_SLICE;
  const int s1 = s0 + STAGES_PER_SLICE;

  // async copy of one 32KB stage (64 K-values x 256 cols) into LDS buffer
  auto issue = [&](int s, unsigned lbase) {
    unsigned goff = (unsigned)(s * 32768 + t * 128);
    unsigned loff = lbase + (unsigned)(t * 128);
#pragma unroll
    for (int j = 0; j < 8; ++j) {
      asm volatile("global_load_async_to_lds_b128 %0, %1, %2"
                   :: "v"(loff + j * 16), "v"(goff + j * 16), "s"(wbase)
                   : "memory");
    }
  };

  issue(s0, ldsb[s0 & 1]);

  f32x8 acc[4] = {};
  for (int s = s0; s < s1; ++s) {
    if (s + 1 < s1) {
      issue(s + 1, ldsb[(s + 1) & 1]);
      asm volatile("s_wait_asynccnt 8" ::: "memory");   // prior stage retired
    } else {
      asm volatile("s_wait_asynccnt 0" ::: "memory");
    }
    __syncthreads();                                    // panel visible to all waves
    const unsigned short* bpanel = &Bsm[s & 1][0];
#pragma unroll
    for (int h = 0; h < 2; ++h) {                       // two 32-K chunks per stage
      int kb = s * 64 + h * 32;
      int patch = kb >> 11;                             // 3x3 tap
      int c0 = kb & (CIN - 1);
      int ky = patch / 3, kx = patch % 3;
      const unsigned short* ap =
          fpad + (((y + ky) * 42) + (x + kx)) * CIN + c0 + half * 8;
      Frag fa;
      fa.q[0] = *(const uint4*)ap;
      fa.q[1] = *(const uint4*)(ap + 16);
      __builtin_prefetch(ap + 64, 0, 1);
#pragma unroll
      for (int q = 0; q < 4; ++q) {
        int cq = nq * 64 + q * 16 + lm;                 // B/D column
        const unsigned short* bp = bpanel + h * 8192 + cq * 32 + half * 8;
        Frag fb;
        fb.q[0] = *(const uint4*)bp;                    // ds_load_b128
        fb.q[1] = *(const uint4*)(bp + 16);
        acc[q] = __builtin_amdgcn_wmma_f32_16x16x32_bf16(
            false, fa.v, false, fb.v, (short)0, acc[q], false, false);
      }
    }
    __syncthreads();                                    // reads done before overwrite
  }

  float* out = xpart + (size_t)kslice * (NPIX * CMID);
#pragma unroll
  for (int q = 0; q < 4; ++q) {
    int col = nq * 64 + q * 16 + lm;
#pragma unroll
    for (int v = 0; v < 8; ++v) {
      int orow = mt * 16 + half * 8 + v;                // D: VGPR v -> M = v + 8*half
      out[orow * CMID + col] = acc[q][v];
    }
  }
}

// ---------- 3b) reduce K-slices + bias + ReLU ----------
__global__ void k_reduce_bias_relu(const float* __restrict__ xpart,
                                   const float* __restrict__ bias,
                                   float* __restrict__ xout) {
  int t = blockIdx.x * blockDim.x + threadIdx.x;
  if (t >= NPIX * CMID) return;
  float s = bias[t & (CMID - 1)];
  s += xpart[t];
  s += xpart[t + 1 * NPIX * CMID];
  s += xpart[t + 2 * NPIX * CMID];
  s += xpart[t + 3 * NPIX * CMID];
  xout[t] = s > 0.f ? s : 0.f;
}

// ---------- 4) 1x1 heads: logits [1600][10], deltas [1600][20] ----------
__global__ void k_heads1x1(const float* __restrict__ xin,
                           const float* __restrict__ cls_w, const float* __restrict__ cls_b,
                           const float* __restrict__ reg_w, const float* __restrict__ reg_b,
                           float* __restrict__ logits, float* __restrict__ deltas) {
  int t = blockIdx.x * blockDim.x + threadIdx.x;
  if (t >= NPIX * 30) return;
  int p = t / 30, o = t % 30;
  const float* xr = xin + p * CMID;
  if (o < 10) {
    float s = cls_b[o];
    for (int c = 0; c < CMID; ++c) s += xr[c] * cls_w[c * 10 + o];
    logits[p * 10 + o] = s;
  } else {
    int o2 = o - 10;
    float s = reg_b[o2];
    for (int c = 0; c < CMID; ++c) s += xr[c] * reg_w[c * 20 + o2];
    deltas[p * 20 + o2] = s;
  }
}

// ---------- 5) anchor decode + softmax score ----------
__global__ void k_anchors(const float* __restrict__ logits,
                          const float* __restrict__ deltas,
                          float* __restrict__ scores, float* __restrict__ boxes) {
  int i = blockIdx.x * blockDim.x + threadIdx.x;
  if (i >= NANCH) return;
  int p = i / 5, a = i % 5;
  const float sc_tab[5] = {32.f, 64.f, 128.f, 256.f, 512.f};
  float l0 = logits[p * 10 + 2 * a], l1 = logits[p * 10 + 2 * a + 1];
  scores[i] = 1.f / (1.f + __expf(l0 - l1));
  float d0 = deltas[p * 20 + 4 * a + 0];
  float d1 = deltas[p * 20 + 4 * a + 1];
  float d2 = deltas[p * 20 + 4 * a + 2];
  float d3 = deltas[p * 20 + 4 * a + 3];
  float wb = sc_tab[a];
  float wdt = __expf(d2) * wb, hgt = __expf(d3) * wb;
  float xc = (float)(p % Ww) + d0;
  float yc = (float)(p / Ww) + d1;
  boxes[i * 4 + 0] = xc - 0.5f * wdt;
  boxes[i * 4 + 1] = yc - 0.5f * hgt;
  boxes[i * 4 + 2] = xc + 0.5f * wdt;
  boxes[i * 4 + 3] = yc + 0.5f * hgt;
}

// ---------- 6) bitonic argsort (descending score) + gather boxes ----------
__global__ __launch_bounds__(1024) void k_sort(const float* __restrict__ scores,
                                               const float* __restrict__ boxes,
                                               float* __restrict__ sboxes) {
  __shared__ float skey[SORTN];   // -score, ascending sort
  __shared__ int   sidx[SORTN];
  int tid = threadIdx.x;
  for (int i = tid; i < SORTN; i += 1024) {
    skey[i] = (i < NANCH) ? -scores[i] : 1e30f;
    sidx[i] = (i < NANCH) ? i : 0;
  }
  for (int k = 2; k <= SORTN; k <<= 1) {
    for (int j = k >> 1; j > 0; j >>= 1) {
      __syncthreads();
      for (int t = tid; t < SORTN / 2; t += 1024) {
        int i = ((t & ~(j - 1)) << 1) | (t & (j - 1));
        int p = i | j;
        bool up = ((i & k) == 0);
        float ki = skey[i], kp = skey[p];
        if ((ki > kp) == up) {
          skey[i] = kp; skey[p] = ki;
          int ti = sidx[i]; sidx[i] = sidx[p]; sidx[p] = ti;
        }
      }
    }
  }
  __syncthreads();
  for (int i = tid; i < NANCH; i += 1024) {
    int s = sidx[i];
    sboxes[i * 4 + 0] = boxes[s * 4 + 0];
    sboxes[i * 4 + 1] = boxes[s * 4 + 1];
    sboxes[i * 4 + 2] = boxes[s * 4 + 2];
    sboxes[i * 4 + 3] = boxes[s * 4 + 3];
  }
}

// ---------- 7) greedy NMS over sorted boxes ----------
__global__ __launch_bounds__(1024) void k_nms(const float* __restrict__ sb,
                                              int* __restrict__ keep_g) {
  __shared__ char keep[NANCH];
  __shared__ float bi[4];
  __shared__ int keepi;
  int tid = threadIdx.x;
  for (int j = tid; j < NANCH; j += 1024) keep[j] = 1;
  for (int i = 0; i < NANCH; ++i) {
    __syncthreads();
    if (tid == 0) {
      keepi = keep[i];
      bi[0] = sb[i * 4 + 0]; bi[1] = sb[i * 4 + 1];
      bi[2] = sb[i * 4 + 2]; bi[3] = sb[i * 4 + 3];
    }
    __syncthreads();
    if (!keepi) continue;
    float x1 = bi[0], y1 = bi[1], x2 = bi[2], y2 = bi[3];
    float ai = (x2 - x1) * (y2 - y1);
    for (int j = i + 1 + tid; j < NANCH; j += 1024) {
      if (!keep[j]) continue;
      float jx1 = sb[j * 4 + 0], jy1 = sb[j * 4 + 1];
      float jx2 = sb[j * 4 + 2], jy2 = sb[j * 4 + 3];
      float iw = fminf(x2, jx2) - fmaxf(x1, jx1);
      float ih = fminf(y2, jy2) - fmaxf(y1, jy1);
      iw = fmaxf(iw, 0.f); ih = fmaxf(ih, 0.f);
      float inter = iw * ih;
      float aj = (jx2 - jx1) * (jy2 - jy1);
      float iou = inter / (ai + aj - inter);
      if (iou >= 0.7f) keep[j] = 0;
    }
  }
  __syncthreads();
  for (int j = tid; j < NANCH; j += 1024) keep_g[j] = keep[j];
}

// ---------- 8) compact kept boxes -> rois (d_out + 2400) ----------
__global__ void k_compact(const float* __restrict__ sb,
                          const int* __restrict__ keep,
                          float* __restrict__ rois) {
  if (threadIdx.x != 0 || blockIdx.x != 0) return;
  int cnt = 0;
  for (int i = 0; i < NANCH && cnt < MAXDET; ++i) {
    if (keep[i]) {
      rois[cnt * 4 + 0] = sb[i * 4 + 0];
      rois[cnt * 4 + 1] = sb[i * 4 + 1];
      rois[cnt * 4 + 2] = sb[i * 4 + 2];
      rois[cnt * 4 + 3] = sb[i * 4 + 3];
      ++cnt;
    }
  }
  for (; cnt < MAXDET; ++cnt) {
    rois[cnt * 4 + 0] = 0.f; rois[cnt * 4 + 1] = 0.f;
    rois[cnt * 4 + 2] = 0.f; rois[cnt * 4 + 3] = 0.f;
  }
}

// ---------- 9) fc1: fc = relu(rois @ fc1_w + fc1_b) ----------
__global__ void k_fc1(const float* __restrict__ rois,
                      const float* __restrict__ w, const float* __restrict__ b,
                      float* __restrict__ fc) {
  int t = blockIdx.x * blockDim.x + threadIdx.x;
  if (t >= MAXDET * 1024) return;
  int r = t >> 10, u = t & 1023;
  float s = b[u];
  s += rois[r * 4 + 0] * w[0 * 1024 + u];
  s += rois[r * 4 + 1] * w[1 * 1024 + u];
  s += rois[r * 4 + 2] * w[2 * 1024 + u];
  s += rois[r * 4 + 3] * w[3 * 1024 + u];
  fc[t] = s > 0.f ? s : 0.f;
}

// ---------- 10) class head (softmax) + box head ----------
__global__ void k_head_out(const float* __restrict__ fc,
                           const float* __restrict__ cw, const float* __restrict__ cb,
                           const float* __restrict__ rw, const float* __restrict__ rb,
                           float* __restrict__ cls_out, float* __restrict__ reg_out) {
  int r = blockIdx.x * blockDim.x + threadIdx.x;
  if (r >= MAXDET) return;
  const float* fr = fc + r * 1024;
  float lg[4];
  for (int o = 0; o < 4; ++o) {
    float s = cb[o];
    for (int u = 0; u < 1024; ++u) s += fr[u] * cw[u * 4 + o];
    lg[o] = s;
  }
  float m = fmaxf(fmaxf(lg[0], lg[1]), fmaxf(lg[2], lg[3]));
  float e0 = __expf(lg[0] - m), e1 = __expf(lg[1] - m);
  float e2 = __expf(lg[2] - m), e3 = __expf(lg[3] - m);
  float inv = 1.f / (e0 + e1 + e2 + e3);
  cls_out[r * 4 + 0] = e0 * inv; cls_out[r * 4 + 1] = e1 * inv;
  cls_out[r * 4 + 2] = e2 * inv; cls_out[r * 4 + 3] = e3 * inv;
  for (int o = 0; o < 4; ++o) {
    float s = rb[o];
    for (int u = 0; u < 1024; ++u) s += fr[u] * rw[u * 4 + o];
    reg_out[r * 4 + o] = s;
  }
}

extern "C" void kernel_launch(void* const* d_in, const int* in_sizes, int n_in,
                              void* d_out, int out_size, void* d_ws, size_t ws_size,
                              hipStream_t stream) {
  const float* features = (const float*)d_in[0];
  const float* conv_w   = (const float*)d_in[1];
  const float* conv_b   = (const float*)d_in[2];
  const float* cls_w    = (const float*)d_in[3];
  const float* cls_b    = (const float*)d_in[4];
  const float* reg_w    = (const float*)d_in[5];
  const float* reg_b    = (const float*)d_in[6];
  const float* fc1_w    = (const float*)d_in[7];
  const float* fc1_b    = (const float*)d_in[8];
  const float* clsh_w   = (const float*)d_in[9];
  const float* clsh_b   = (const float*)d_in[10];
  const float* regh_w   = (const float*)d_in[11];
  const float* regh_b   = (const float*)d_in[12];

  char* ws = (char*)d_ws;
  unsigned short* fpad   = (unsigned short*)(ws + OFS_FEATPAD);
  unsigned short* wpack  = (unsigned short*)(ws + OFS_WPACK);
  float* xmid   = (float*)(ws + OFS_X);
  float* logits = (float*)(ws + OFS_LOGITS);
  float* deltas = (float*)(ws + OFS_DELTAS);
  float* scores = (float*)(ws + OFS_SCORES);
  float* boxes  = (float*)(ws + OFS_BOXES);
  float* sboxes = (float*)(ws + OFS_SBOXES);
  int*   keep   = (int*)(ws + OFS_KEEP);
  float* fc     = (float*)(ws + OFS_FC);
  float* xpart  = (float*)(ws + OFS_XPART);

  float* cls_out = (float*)d_out;             // [300][4]
  float* reg_out = cls_out + MAXDET * 4;      // [300][4]
  float* rois    = reg_out + MAXDET * 4;      // [300][4]

  k_pad_features<<<(42 * 42 * CIN + 255) / 256, 256, 0, stream>>>(features, fpad);
  k_pack_weights<<<(KDIM * CMID + 255) / 256, 256, 0, stream>>>(conv_w, wpack);
  k_conv3x3_wmma<<<50 * KSLICES, 256, 0, stream>>>(fpad, wpack, xpart);
  k_reduce_bias_relu<<<(NPIX * CMID + 255) / 256, 256, 0, stream>>>(xpart, conv_b, xmid);
  k_heads1x1<<<(NPIX * 30 + 255) / 256, 256, 0, stream>>>(xmid, cls_w, cls_b,
                                                          reg_w, reg_b, logits, deltas);
  k_anchors<<<(NANCH + 255) / 256, 256, 0, stream>>>(logits, deltas, scores, boxes);
  k_sort<<<1, 1024, 0, stream>>>(scores, boxes, sboxes);
  k_nms<<<1, 1024, 0, stream>>>(sboxes, keep);
  k_compact<<<1, 64, 0, stream>>>(sboxes, keep, rois);
  k_fc1<<<(MAXDET * 1024 + 255) / 256, 256, 0, stream>>>(rois, fc1_w, fc1_b, fc);
  k_head_out<<<(MAXDET + 63) / 64, 64, 0, stream>>>(fc, clsh_w, clsh_b,
                                                    regh_w, regh_b, cls_out, reg_out);
}
